// CoAttentionFusion_20641612825100
// MI455X (gfx1250) — matrix operand order repaired and checked by reading, not compiled, not run on hardware
//
#include <hip/hip_runtime.h>

// ---------------------------------------------------------------------------
// CoAttention fusion for MI455X (gfx1250, wave32, WMMA).
// Templated bf16 WMMA GEMM: v_wmma_f32_16x16x32_bf16, 128x128x32 tiles,
// 8 waves/block, 4x2 WMMA tiles per wave. Double-buffered LDS, prefetch
// pinned before compute via sched_barrier, async-to-LDS where available.
// Weights are pre-transposed so their staging is pure b128 copies.
// ---------------------------------------------------------------------------

typedef __attribute__((ext_vector_type(16))) __bf16 v16bf;
typedef __attribute__((ext_vector_type(8)))  __bf16 v8bf;
typedef __attribute__((ext_vector_type(8)))  float  v8f;
typedef __attribute__((ext_vector_type(4)))  int    vi4;

#define BM 128
#define BN 128
#define BKK 32
#define KPAD 8
#define LDSLD (BKK + KPAD)   // 40 elem (80B) row stride: conflict-free b128 frag reads

#define EPI_BF16      0
#define EPI_TANH_MASK 1
#define EPI_RELU_AMAX 2
#define EPI_BIAS_RELU 3

#if defined(__HIP_DEVICE_COMPILE__) && \
    __has_builtin(__builtin_amdgcn_global_load_async_to_lds_b128) && \
    __has_builtin(__builtin_amdgcn_s_wait_asynccnt)
#define HAVE_ASYNC 1
#else
#define HAVE_ASYNC 0
#endif

#if defined(__HIP_DEVICE_COMPILE__) && __has_builtin(__builtin_amdgcn_sched_barrier)
#define SCHED_FENCE() __builtin_amdgcn_sched_barrier(0)
#else
#define SCHED_FENCE()
#endif

#if HAVE_ASYNC
__device__ __forceinline__ void async_cp16(const void* g, void* l) {
  __builtin_amdgcn_global_load_async_to_lds_b128(
      (__attribute__((address_space(1))) vi4*)g,
      (__attribute__((address_space(3))) vi4*)l, 0, 0);
}
#endif

struct GemmArgs {
  const __bf16* A;
  const __bf16* B;
  void*         C;
  const float*  maskM;
  const float*  maskN;
  const float*  bias;
  int K;
  int lda, ldb, ldc;
  int zdiv;                       // z -> (b1 = z/zdiv, b2 = z%zdiv)
  long long sA1, sA2, sB1, sB2, sC1, sC2;
  int maskMstride, maskNstride;
};

// TA: A memory is K x M (needs transpose-scatter).  TB: B memory is N x K (direct copy).
template <bool TA, bool TB, int EPI>
__global__ __launch_bounds__(256) void gemm_wmma(GemmArgs g) {
  __shared__ __attribute__((aligned(16))) __bf16 As[2][BM][LDSLD];
  __shared__ __attribute__((aligned(16))) __bf16 Bs[2][BN][LDSLD];

  const int tid  = threadIdx.x;
  const int wid  = tid >> 5;
  const int lane = tid & 31;
  const int half = lane >> 4;
  const int ln16 = lane & 15;
  const int wr   = wid >> 2;   // 0..1
  const int wc   = wid & 3;    // 0..3

  const int m0 = blockIdx.y * BM;
  const int n0 = blockIdx.x * BN;
  const int z  = blockIdx.z;
  const int b1 = z / g.zdiv;
  const int b2 = z % g.zdiv;

  const __bf16* Ab = g.A + b1 * g.sA1 + b2 * g.sA2;
  const __bf16* Bb = g.B + b1 * g.sB1 + b2 * g.sB2;

  v8f acc[4][2];
#pragma unroll
  for (int ms = 0; ms < 4; ++ms)
#pragma unroll
    for (int ns = 0; ns < 2; ++ns)
#pragma unroll
      for (int e = 0; e < 8; ++e) acc[ms][ns][e] = 0.f;

  // ---- prefetch: issue global loads (async directly into LDS where direct-copy)
  auto prefetchA = [&](int k0, int buf, v8bf ra[2]) {
#pragma unroll
    for (int it = 0; it < 2; ++it) {
      int idx = it * 256 + tid;
      if (!TA) {
        int row = idx >> 2, col = (idx & 3) * 8;
        const __bf16* gp = Ab + (long long)(m0 + row) * g.lda + k0 + col;
#if HAVE_ASYNC
        async_cp16(gp, &As[buf][row][col]);
#else
        ra[it] = *(const v8bf*)gp;
#endif
      } else {
        int krow = idx >> 4, mcol = (idx & 15) * 8;
        ra[it] = *(const v8bf*)(Ab + (long long)(k0 + krow) * g.lda + m0 + mcol);
      }
    }
  };
  auto prefetchB = [&](int k0, int buf, v8bf rb[2]) {
#pragma unroll
    for (int it = 0; it < 2; ++it) {
      int idx = it * 256 + tid;
      if (TB) {
        int nrow = idx >> 2, kcol = (idx & 3) * 8;
        const __bf16* gp = Bb + (long long)(n0 + nrow) * g.ldb + k0 + kcol;
#if HAVE_ASYNC
        async_cp16(gp, &Bs[buf][nrow][kcol]);
#else
        rb[it] = *(const v8bf*)gp;
#endif
      } else {
        int krow = idx >> 4, ncol = (idx & 15) * 8;
        rb[it] = *(const v8bf*)(Bb + (long long)(k0 + krow) * g.ldb + n0 + ncol);
      }
    }
  };
  // ---- commit: registers -> LDS (no-op for async direct paths)
  auto commitA = [&](int buf, v8bf ra[2]) {
#pragma unroll
    for (int it = 0; it < 2; ++it) {
      int idx = it * 256 + tid;
      if (!TA) {
#if !HAVE_ASYNC
        int row = idx >> 2, col = (idx & 3) * 8;
        *(v8bf*)&As[buf][row][col] = ra[it];
#endif
      } else {
        int krow = idx >> 4, mcol = (idx & 15) * 8;
#pragma unroll
        for (int j = 0; j < 8; ++j) As[buf][mcol + j][krow] = ra[it][j];
      }
    }
  };
  auto commitB = [&](int buf, v8bf rb[2]) {
#pragma unroll
    for (int it = 0; it < 2; ++it) {
      int idx = it * 256 + tid;
      if (TB) {
#if !HAVE_ASYNC
        int nrow = idx >> 2, kcol = (idx & 3) * 8;
        *(v8bf*)&Bs[buf][nrow][kcol] = rb[it];
#endif
      } else {
        int krow = idx >> 4, ncol = (idx & 15) * 8;
#pragma unroll
        for (int j = 0; j < 8; ++j) Bs[buf][ncol + j][krow] = rb[it][j];
      }
    }
  };
  // ---- LDS fragments + 8 WMMAs (ISA 7.12.2 layouts)
  auto compute = [&](int buf) {
    v16bf afrag[4], bfrag[2];
#pragma unroll
    for (int ms = 0; ms < 4; ++ms) {
      int mrow = wr * 64 + ms * 16 + ln16;
      v8bf lo = *(const v8bf*)&As[buf][mrow][8 * half];
      v8bf hi = *(const v8bf*)&As[buf][mrow][16 + 8 * half];
#pragma unroll
      for (int e = 0; e < 8; ++e) { afrag[ms][e] = lo[e]; afrag[ms][8 + e] = hi[e]; }
    }
#pragma unroll
    for (int ns = 0; ns < 2; ++ns) {
      int nrow = wc * 32 + ns * 16 + ln16;
      v8bf lo = *(const v8bf*)&Bs[buf][nrow][8 * half];
      v8bf hi = *(const v8bf*)&Bs[buf][nrow][16 + 8 * half];
#pragma unroll
      for (int e = 0; e < 8; ++e) { bfrag[ns][e] = lo[e]; bfrag[ns][8 + e] = hi[e]; }
    }
#pragma unroll
    for (int ms = 0; ms < 4; ++ms)
#pragma unroll
      for (int ns = 0; ns < 2; ++ns)
        acc[ms][ns] = __builtin_amdgcn_wmma_f32_16x16x32_bf16(
            false, afrag[ms], false, bfrag[ns], (short)0, acc[ms][ns], false, false);
  };

  // ---- pipeline: issue k+1, compute k, commit k+1 into other buffer, 1 barrier
  v8bf ra[2], rb[2];
  prefetchA(0, 0, ra);
  prefetchB(0, 0, rb);
  commitA(0, ra);
  commitB(0, rb);
#if HAVE_ASYNC
  __builtin_amdgcn_s_wait_asynccnt(0);
#endif
  __syncthreads();
  int cur = 0;
  for (int k0 = BKK; k0 < g.K; k0 += BKK) {
    prefetchA(k0, cur ^ 1, ra);       // in flight during WMMAs
    prefetchB(k0, cur ^ 1, rb);
    SCHED_FENCE();                    // keep load issue above the compute block
    compute(cur);
    commitA(cur ^ 1, ra);
    commitB(cur ^ 1, rb);
#if HAVE_ASYNC
    __builtin_amdgcn_s_wait_asynccnt(0);
#endif
    __syncthreads();
    cur ^= 1;
  }
  compute(cur);

  // ---- epilogue: C reg r -> M = r + 8*half, N = ln16
  const long long coff = b1 * g.sC1 + b2 * g.sC2;
#pragma unroll
  for (int ms = 0; ms < 4; ++ms) {
#pragma unroll
    for (int ns = 0; ns < 2; ++ns) {
      const int gn = n0 + wc * 32 + ns * 16 + ln16;
#pragma unroll
      for (int r = 0; r < 8; ++r) {
        const int gm = m0 + wr * 64 + ms * 16 + 8 * half + r;
        float v = acc[ms][ns][r];
        const long long cidx = coff + (long long)gm * g.ldc + gn;
        if (EPI == EPI_BF16) {
          ((__bf16*)g.C)[cidx] = (__bf16)v;
        } else if (EPI == EPI_TANH_MASK) {
          float mm = g.maskM[b1 * g.maskMstride + gm];
          float mn = g.maskN[b1 * g.maskNstride + gn];
          ((__bf16*)g.C)[cidx] = (__bf16)(tanhf(v) * mm * mn);
        } else if (EPI == EPI_RELU_AMAX) {
          float rv = v > 0.f ? v : 0.f;            // >=0: uint max == float max
          atomicMax((unsigned int*)g.C + cidx, __float_as_uint(rv));
        } else { // EPI_BIAS_RELU
          float rv = v + g.bias[gn];
          ((float*)g.C)[cidx] = rv > 0.f ? rv : 0.f;
        }
      }
    }
  }
}

__global__ void f32_to_bf16(const float* __restrict__ in, __bf16* __restrict__ out, int n) {
  int i = blockIdx.x * blockDim.x + threadIdx.x;
  if (i < n) out[i] = (__bf16)in[i];
}

// out[z][n][k] = bf16(in[z][k][n]) : per-batch transpose (K x N -> N x K)
__global__ void f32_to_bf16_T(const float* __restrict__ in, __bf16* __restrict__ out,
                              int K, int N) {
  long long base = (long long)blockIdx.z * K * N;
  int idx = blockIdx.x * blockDim.x + threadIdx.x;
  if (idx >= K * N) return;
  int n = idx / K, k = idx % K;                 // consecutive idx -> consecutive k (coalesced out)
  out[base + idx] = (__bf16)in[base + (long long)k * N + n];
}

// cat[row, 0:512] = bf16(x[row,:]); cat[row, 512:640] = bf16(pool[row,:])
__global__ void concat_bf16(const float* __restrict__ x, const float* __restrict__ pool,
                            __bf16* __restrict__ out, int total) {
  int idx = blockIdx.x * blockDim.x + threadIdx.x;
  if (idx >= total) return;
  int row = idx / 640, col = idx % 640;
  float v = (col < 512) ? x[row * 512 + col] : pool[row * 128 + (col - 512)];
  out[idx] = (__bf16)v;
}

extern "C" void kernel_launch(void* const* d_in, const int* in_sizes, int n_in,
                              void* d_out, int out_size, void* d_ws, size_t ws_size,
                              hipStream_t stream) {
  constexpr int Bz = 16, L1 = 512, L2 = 512, Dd = 512, Hh = 8, INNER = 1024, HDd = 128;

  const float* primary   = (const float*)d_in[0];
  const float* secondary = (const float*)d_in[1];
  const float* maskP     = (const float*)d_in[2];
  const float* maskS     = (const float*)d_in[3];
  const float* W_aff     = (const float*)d_in[4];
  const float* W_p       = (const float*)d_in[5];
  const float* W_s       = (const float*)d_in[6];
  const float* W_fp      = (const float*)d_in[7];
  const float* b_fp      = (const float*)d_in[8];
  const float* W_fs      = (const float*)d_in[9];
  const float* b_fs      = (const float*)d_in[10];

  char* ws = (char*)d_ws;
  size_t o = 0;
  auto wsalloc = [&](size_t bytes) { size_t r = o; o += (bytes + 255) & ~(size_t)255; return r; };

  __bf16* P_bf    = (__bf16*)(ws + wsalloc((size_t)Bz * L1 * Dd * 2));
  __bf16* S_bf    = (__bf16*)(ws + wsalloc((size_t)Bz * L2 * Dd * 2));
  __bf16* WaffT   = (__bf16*)(ws + wsalloc((size_t)Hh * Dd * Dd * 2));       // (h, N=D, K=D)
  __bf16* WpT     = (__bf16*)(ws + wsalloc((size_t)INNER * Dd * 2));         // (N=INNER, K=D)
  __bf16* WsT     = (__bf16*)(ws + wsalloc((size_t)INNER * Dd * 2));
  __bf16* WfpT    = (__bf16*)(ws + wsalloc((size_t)Dd * (Dd + HDd) * 2));    // (N=D, K=640)
  __bf16* WfsT    = (__bf16*)(ws + wsalloc((size_t)Dd * (Dd + HDd) * 2));
  __bf16* T_bf    = (__bf16*)(ws + wsalloc((size_t)Bz * L1 * Dd * 2));       // reused per head
  __bf16* aff_bf  = (__bf16*)(ws + wsalloc((size_t)Bz * Hh * L1 * L2 * 2));
  __bf16* pp_bf   = (__bf16*)(ws + wsalloc((size_t)Bz * L1 * INNER * 2));
  __bf16* ps_bf   = (__bf16*)(ws + wsalloc((size_t)Bz * L2 * INNER * 2));
  float*  pool_p  = (float*) (ws + wsalloc((size_t)Bz * L2 * HDd * 4));
  float*  pool_s  = (float*) (ws + wsalloc((size_t)Bz * L1 * HDd * 4));
  __bf16* catP    = (__bf16*)(ws + wsalloc((size_t)Bz * L1 * (Dd + HDd) * 2));
  __bf16* catS    = (__bf16*)(ws + wsalloc((size_t)Bz * L2 * (Dd + HDd) * 2));

  auto cvt = [&](const float* src, __bf16* dst, int n) {
    f32_to_bf16<<<(n + 255) / 256, 256, 0, stream>>>(src, dst, n);
  };
  auto cvtT = [&](const float* src, __bf16* dst, int K, int N, int batch) {
    f32_to_bf16_T<<<dim3((K * N + 255) / 256, 1, batch), 256, 0, stream>>>(src, dst, K, N);
  };
  cvt(primary,   P_bf, Bz * L1 * Dd);
  cvt(secondary, S_bf, Bz * L2 * Dd);
  cvtT(W_aff, WaffT, Dd, Dd, Hh);
  cvtT(W_p,   WpT,   Dd, INNER, 1);
  cvtT(W_s,   WsT,   Dd, INNER, 1);
  cvtT(W_fp,  WfpT,  Dd + HDd, Dd, 1);
  cvtT(W_fs,  WfsT,  Dd + HDd, Dd, 1);

  (void)hipMemsetAsync(pool_p, 0, (size_t)Bz * L2 * HDd * 4, stream);
  (void)hipMemsetAsync(pool_s, 0, (size_t)Bz * L1 * HDd * 4, stream);

  // pp = P @ W_p (B uses transposed weights: direct-copy staging)
  GemmArgs g = {};
  g.zdiv = 1;
  g.A = P_bf; g.B = WpT; g.C = pp_bf;
  g.K = Dd; g.lda = Dd; g.ldb = Dd; g.ldc = INNER;
  gemm_wmma<false, true, EPI_BF16><<<dim3(INNER / BN, (Bz * L1) / BM, 1), 256, 0, stream>>>(g);
  g.A = S_bf; g.B = WsT; g.C = ps_bf;
  gemm_wmma<false, true, EPI_BF16><<<dim3(INNER / BN, (Bz * L2) / BM, 1), 256, 0, stream>>>(g);

  for (int h = 0; h < Hh; ++h) {
    // T[b] = P[b] @ W_aff[h]
    GemmArgs g1 = {};
    g1.A = P_bf; g1.B = WaffT + (size_t)h * Dd * Dd; g1.C = T_bf;
    g1.K = Dd; g1.lda = Dd; g1.ldb = Dd; g1.ldc = Dd;
    g1.zdiv = 1; g1.sA1 = (long long)L1 * Dd; g1.sC1 = (long long)L1 * Dd;
    gemm_wmma<false, true, EPI_BF16><<<dim3(Dd / BN, L1 / BM, Bz), 256, 0, stream>>>(g1);

    // aff[b,h] = tanh(T[b] @ S[b]^T) * mask  (S is N x K row-major: direct copy)
    GemmArgs g2 = {};
    g2.A = T_bf; g2.B = S_bf; g2.C = aff_bf + (size_t)h * L1 * L2;
    g2.maskM = maskP; g2.maskN = maskS; g2.maskMstride = L1; g2.maskNstride = L2;
    g2.K = Dd; g2.lda = Dd; g2.ldb = Dd; g2.ldc = L2;
    g2.zdiv = 1; g2.sA1 = (long long)L1 * Dd; g2.sB1 = (long long)L2 * Dd;
    g2.sC1 = (long long)Hh * L1 * L2;
    gemm_wmma<false, true, EPI_TANH_MASK><<<dim3(L2 / BN, L1 / BM, Bz), 256, 0, stream>>>(g2);
  }

  // pool_s[b] = max_h relu(aff[b,h] @ ps[b,:,h,:])   z = b*H + h
  GemmArgs gw = {};
  gw.A = aff_bf; gw.B = ps_bf; gw.C = pool_s;
  gw.K = L2; gw.lda = L2; gw.ldb = INNER; gw.ldc = HDd;
  gw.zdiv = Hh;
  gw.sA1 = (long long)Hh * L1 * L2; gw.sA2 = (long long)L1 * L2;
  gw.sB1 = (long long)L2 * INNER;   gw.sB2 = HDd;
  gw.sC1 = (long long)L1 * HDd;
  gemm_wmma<false, false, EPI_RELU_AMAX><<<dim3(1, L1 / BM, Bz * Hh), 256, 0, stream>>>(gw);

  // pool_p[b] = max_h relu(aff[b,h]^T @ pp[b,:,h,:])  (A mem K x M: scatter staging)
  GemmArgs gv = {};
  gv.A = aff_bf; gv.B = pp_bf; gv.C = pool_p;
  gv.K = L1; gv.lda = L2; gv.ldb = INNER; gv.ldc = HDd;
  gv.zdiv = Hh;
  gv.sA1 = (long long)Hh * L1 * L2; gv.sA2 = (long long)L1 * L2;
  gv.sB1 = (long long)L1 * INNER;   gv.sB2 = HDd;
  gv.sC1 = (long long)L2 * HDd;
  gemm_wmma<true, false, EPI_RELU_AMAX><<<dim3(1, L2 / BM, Bz * Hh), 256, 0, stream>>>(gv);

  {
    int totP = Bz * L1 * (Dd + HDd);
    concat_bf16<<<(totP + 255) / 256, 256, 0, stream>>>(primary, pool_s, catP, totP);
    int totS = Bz * L2 * (Dd + HDd);
    concat_bf16<<<(totS + 255) / 256, 256, 0, stream>>>(secondary, pool_p, catS, totS);
  }

  // out_p = relu(catP @ W_fp + b_fp) ; out_s = relu(catS @ W_fs + b_fs)
  float* out_p = (float*)d_out;
  float* out_s = out_p + (size_t)Bz * L1 * Dd;
  GemmArgs go = {};
  go.A = catP; go.B = WfpT; go.C = out_p; go.bias = b_fp;
  go.K = Dd + HDd; go.lda = Dd + HDd; go.ldb = Dd + HDd; go.ldc = Dd;
  go.zdiv = 1;
  gemm_wmma<false, true, EPI_BIAS_RELU><<<dim3(Dd / BN, (Bz * L1) / BM, 1), 256, 0, stream>>>(go);
  go.A = catS; go.B = WfsT; go.C = out_s; go.bias = b_fs;
  gemm_wmma<false, true, EPI_BIAS_RELU><<<dim3(Dd / BN, (Bz * L2) / BM, 1), 256, 0, stream>>>(go);

  (void)in_sizes; (void)n_in; (void)out_size; (void)ws_size;
}
